// GatedDeltaNetLayer_6863357739623
// MI455X (gfx1250) — matrix operand order repaired and checked by reading, not compiled, not run on hardware
//
#include <hip/hip_runtime.h>
#include <hip/hip_bf16.h>
#include <math.h>

// ---------------- problem constants (from reference) ----------------
namespace {
constexpr int B_   = 4;
constexpr int T_   = 512;
constexpr int C_   = 1024;
constexpr int H_   = 8;
constexpr int DK   = 96;
constexpr int DV   = 192;
constexpr int KC   = 4;                 // conv kernel width
constexpr int QKD  = H_ * DK;           // 768
constexpr int VD   = H_ * DV;           // 1536
constexpr int RSZ  = H_ * DK * DV;      // 147456
constexpr int CACHE_ = RSZ + 2 * QKD * KC + VD * KC; // 159744
constexpr int BT   = B_ * T_;           // 2048
constexpr int OFF_CQ = RSZ;             // 147456
constexpr int OFF_CK = RSZ + QKD * KC;  // 150528
constexpr int OFF_CV = RSZ + 2 * QKD * KC; // 153600
constexpr int CH   = 8;                 // recurrence time-chunk (TDM staged)
}

typedef __bf16 v16bf __attribute__((ext_vector_type(16)));
typedef __bf16 v8bf  __attribute__((ext_vector_type(8)));
typedef float  v8f   __attribute__((ext_vector_type(8)));
typedef unsigned int u32x4 __attribute__((ext_vector_type(4)));
typedef int i32x4 __attribute__((ext_vector_type(4)));
typedef int i32x8 __attribute__((ext_vector_type(8)));

#if defined(__HIP_DEVICE_COMPILE__) && __has_builtin(__builtin_amdgcn_tensor_load_to_lds)
#define HAVE_TDM 1
#else
#define HAVE_TDM 0
#endif

#if HAVE_TDM
// Issue one TDM 2D tile load: `lines` rows of `w` f32 elements, rows spaced
// `strideElems` apart in global memory, packed contiguously at LDS byte
// offset `ldsOff`. D# encoding per CDNA5 ISA 8.3/8.4.
__device__ __forceinline__ void tdm_load_2d(const float* gsrc, unsigned ldsOff,
                                            int w, int lines, long strideElems) {
    unsigned long long ga = (unsigned long long)(uintptr_t)gsrc;
    u32x4 g0;
    g0[0] = 1u;                                            // count=1 (valid user D#)
    g0[1] = ldsOff;                                        // lds_addr [63:32]
    g0[2] = (unsigned)(ga & 0xffffffffu);                  // global_addr [95:64]
    g0[3] = (unsigned)((ga >> 32) & 0x1ffffffu)            // global_addr [120:96]
          | (2u << 30);                                    // type=2 ("image") [127:126]
    i32x8 g1;
    g1[0] = (2 << 16);                                     // data_size=2 -> 4 bytes
    g1[1] = (w & 0xffff) << 16;                            // tensor_dim0 lo -> [63:48]
    g1[2] = ((w >> 16) & 0xffff) | ((lines & 0xffff) << 16); // tensor_dim0 hi | tensor_dim1 lo
    g1[3] = ((lines >> 16) & 0xffff) | ((w & 0xffff) << 16); // tensor_dim1 hi | tile_dim0
    g1[4] = (lines & 0xffff);                              // tile_dim1 (tile_dim2=0)
    g1[5] = (int)(strideElems & 0xffffffff);               // tensor_dim0_stride lo32
    g1[6] = (int)((strideElems >> 32) & 0xffff);           // tensor_dim0_stride hi16
    g1[7] = 0;
    i32x4 z4 = {0, 0, 0, 0};
#if defined(__clang_major__) && (__clang_major__ >= 23)
    i32x8 z8 = {0, 0, 0, 0, 0, 0, 0, 0};
    __builtin_amdgcn_tensor_load_to_lds(g0, g1, z4, z4, z8, 0);
#else
    __builtin_amdgcn_tensor_load_to_lds(g0, g1, z4, z4, 0);
#endif
}
__device__ __forceinline__ unsigned lds_off_of(const void* p) {
    // LDS aperture flat address: low 32 bits are the LDS byte offset
    return (unsigned)(uintptr_t)p;
}
#endif

// ---------------- fp32 -> bf16 convert ----------------
__global__ void cvt_bf16_kernel(const float* __restrict__ X, __bf16* __restrict__ Y, int n) {
    int i = blockIdx.x * blockDim.x + threadIdx.x;
    if (i < n) Y[i] = (__bf16)X[i];
}

// W [Kd, N] row-major -> WT [N, Kd] row-major, bf16
__global__ void transpose_bf16_kernel(const float* __restrict__ W, __bf16* __restrict__ WT,
                                      int Kd, int N) {
    int idx = blockIdx.x * blockDim.x + threadIdx.x;
    if (idx >= Kd * N) return;
    int n = idx / Kd, c = idx % Kd;
    WT[idx] = (__bf16)W[(size_t)c * N + n];
}

// ---------------- WMMA bf16 GEMM, 32x64 register-blocked per wave ----------------
// Layouts as used here (A/B in contiguous k order, same for both):
//  A 16x32 bf16: lanes 0-15 row M=l,   K = 0..15;           lanes 16-31 row M=l-16, K = 16..31
//  B 32x16 bf16: lanes 0-15 col N=l,   K = 0..15;           lanes 16-31 col N=l-16, K = 16..31
//  D 16x16 f32 : vgpr r: lanes 0-15 -> (M=r, N=l); lanes 16-31 -> (M=8+r, N=l-16)
__global__ __launch_bounds__(256)
void wmma_gemm_bf16_kernel(const __bf16* __restrict__ A,   // [M, Kd]
                           const __bf16* __restrict__ BTm, // [N, Kd]  (B transposed)
                           float* __restrict__ C,          // [M, N]
                           int M, int N, int Kd) {
    const int lane = threadIdx.x & 31;
    const int wave = threadIdx.x >> 5;
    const int nMacs = N >> 6;                       // 64-column macro tiles
    long gw = (long)blockIdx.x * (blockDim.x >> 5) + wave;
    if (gw >= (long)(M >> 5) * nMacs) return;
    const int nMac = (int)(gw % nMacs);
    const int mMac = (int)(gw / nMacs);

    const int aKoff = (lane < 16) ? 0 : 16;
    const int bKoff = (lane < 16) ? 0 : 16;
    const int mBase = mMac << 5;
    const int nBase = nMac << 6;

    const __bf16* aRow0 = A + (size_t)(mBase + (lane & 15)) * Kd + aKoff;
    const __bf16* aRow1 = aRow0 + (size_t)16 * Kd;
    const __bf16* bCol  = BTm + (size_t)(nBase + (lane & 15)) * Kd + bKoff;
    const size_t bStep = (size_t)16 * Kd;

    v8f acc[2][4] = {};
    for (int k0 = 0; k0 < Kd; k0 += 32) {
        v8bf a0lo = *(const v8bf*)(aRow0 + k0);
        v8bf a0hi = *(const v8bf*)(aRow0 + k0 + 8);
        v8bf a1lo = *(const v8bf*)(aRow1 + k0);
        v8bf a1hi = *(const v8bf*)(aRow1 + k0 + 8);
        v16bf a0 = __builtin_shufflevector(a0lo, a0hi, 0,1,2,3,4,5,6,7,8,9,10,11,12,13,14,15);
        v16bf a1 = __builtin_shufflevector(a1lo, a1hi, 0,1,2,3,4,5,6,7,8,9,10,11,12,13,14,15);
        v16bf b0 = *(const v16bf*)(bCol + k0);
        v16bf b1 = *(const v16bf*)(bCol + bStep + k0);
        v16bf b2 = *(const v16bf*)(bCol + 2 * bStep + k0);
        v16bf b3 = *(const v16bf*)(bCol + 3 * bStep + k0);
        __builtin_prefetch(aRow0 + k0 + 128, 0, 0);
        __builtin_prefetch(bCol + k0 + 128, 0, 0);
        acc[0][0] = __builtin_amdgcn_wmma_f32_16x16x32_bf16(false, a0, false, b0, (short)0, acc[0][0], false, false);
        acc[0][1] = __builtin_amdgcn_wmma_f32_16x16x32_bf16(false, a0, false, b1, (short)0, acc[0][1], false, false);
        acc[0][2] = __builtin_amdgcn_wmma_f32_16x16x32_bf16(false, a0, false, b2, (short)0, acc[0][2], false, false);
        acc[0][3] = __builtin_amdgcn_wmma_f32_16x16x32_bf16(false, a0, false, b3, (short)0, acc[0][3], false, false);
        acc[1][0] = __builtin_amdgcn_wmma_f32_16x16x32_bf16(false, a1, false, b0, (short)0, acc[1][0], false, false);
        acc[1][1] = __builtin_amdgcn_wmma_f32_16x16x32_bf16(false, a1, false, b1, (short)0, acc[1][1], false, false);
        acc[1][2] = __builtin_amdgcn_wmma_f32_16x16x32_bf16(false, a1, false, b2, (short)0, acc[1][2], false, false);
        acc[1][3] = __builtin_amdgcn_wmma_f32_16x16x32_bf16(false, a1, false, b3, (short)0, acc[1][3], false, false);
    }

    const int rBase = (lane < 16) ? 0 : 8;
    const int nCol0 = nBase + (lane & 15);
#pragma unroll
    for (int i = 0; i < 2; i++) {
#pragma unroll
        for (int j = 0; j < 4; j++) {
#pragma unroll
            for (int r = 0; r < 8; r++) {
                C[(size_t)(mBase + 16 * i + rBase + r) * N + nCol0 + 16 * j] = acc[i][j][r];
            }
        }
    }
}

// ---------------- beta / g head: one wave per (bt, h) ----------------
__global__ void ab_head_kernel(const float* __restrict__ x,  // [BT, C]
                               const float* __restrict__ Wa, // [C, H]
                               const float* __restrict__ Wb, // [C, H]
                               const float* __restrict__ dtb,
                               const float* __restrict__ Alog,
                               float* __restrict__ g,        // [BT, H]
                               float* __restrict__ beta) {   // [BT, H]
    int gw = (blockIdx.x * blockDim.x + threadIdx.x) >> 5;
    int lane = threadIdx.x & 31;
    if (gw >= BT * H_) return;
    int bt = gw / H_, h = gw % H_;
    const float* xr = x + (size_t)bt * C_;
    float sa = 0.f, sb = 0.f;
    for (int c = lane; c < C_; c += 32) {
        float xv = xr[c];
        sa = fmaf(xv, Wa[c * H_ + h], sa);
        sb = fmaf(xv, Wb[c * H_ + h], sb);
    }
    for (int m = 16; m; m >>= 1) { sa += __shfl_xor(sa, m, 32); sb += __shfl_xor(sb, m, 32); }
    if (lane == 0) {
        float a  = sa + dtb[h];
        float sp = (a > 20.f) ? a : log1pf(__expf(a));
        g[gw]    = -__expf(Alog[h]) * sp;
        beta[gw] = 1.f / (1.f + __expf(-sb));
    }
}

// ---------------- in-place depthwise causal conv (K=4) + SiLU, emits new conv cache ----------------
__global__ void conv_silu_kernel(float* __restrict__ buf,        // [B, T, D] in/out
                                 const float* __restrict__ w,    // [D, K]
                                 const float* __restrict__ state_in,
                                 float* __restrict__ state_out,
                                 int D, int cacheOff) {
    int gid = blockIdx.x * blockDim.x + threadIdx.x;
    if (gid >= B_ * D) return;
    int b = gid / D, d = gid % D;
    const float* cin = state_in + (size_t)b * CACHE_ + cacheOff + (size_t)d * KC;
    float w0 = w[d * KC + 0], w1 = w[d * KC + 1], w2 = w[d * KC + 2], w3 = w[d * KC + 3];
    float x0 = cin[1], x1 = cin[2], x2 = cin[3], x3;
    float* col = buf + (size_t)b * T_ * D + d;
    float* cout = state_out + (size_t)b * CACHE_ + cacheOff + (size_t)d * KC;
    for (int t = 0; t < T_; t++) {
        x3 = col[(size_t)t * D];
        if (t == T_ - 1) { cout[0] = x0; cout[1] = x1; cout[2] = x2; cout[3] = x3; }
        float y = w0 * x0 + w1 * x1 + w2 * x2 + w3 * x3;
        col[(size_t)t * D] = y / (1.f + __expf(-y));   // SiLU, in place
        x0 = x1; x1 = x2; x2 = x3;
    }
}

// ---------------- L2 norm over head dim (96) : one wave per (b,t,h) ----------------
__global__ void l2norm96_kernel(float* __restrict__ buf) {   // [BT*H, 96]
    int gw = (blockIdx.x * blockDim.x + threadIdx.x) >> 5;
    int lane = threadIdx.x & 31;
    if (gw >= BT * H_) return;
    float* p = buf + (size_t)gw * DK;
    float x0 = p[lane], x1 = p[lane + 32], x2 = p[lane + 64];
    float ss = x0 * x0 + x1 * x1 + x2 * x2;
    for (int m = 16; m; m >>= 1) ss += __shfl_xor(ss, m, 32);
    float r = rsqrtf(ss + 1e-6f);
    p[lane] = x0 * r; p[lane + 32] = x1 * r; p[lane + 64] = x2 * r;
}

// ---------------- gated delta-rule recurrence ----------------
// One block per (b,h); S[96,192] lives in VGPRs (thread vc owns column vc).
// k/q rows for CH=8 timesteps are staged in LDS double buffers; on gfx1250
// the staging is done by the Tensor Data Mover (tensor_load_to_lds, waited
// with s_wait_tensorcnt) issued by wave 0 one chunk ahead of compute.
__global__ __launch_bounds__(192)
void recurrence_kernel(const float* __restrict__ q,    // [B,T,H*DK]
                       const float* __restrict__ k,    // [B,T,H*DK]
                       const float* __restrict__ v,    // [B,T,H*DV]
                       const float* __restrict__ g,    // [B,T,H]
                       const float* __restrict__ beta, // [B,T,H]
                       const float* __restrict__ state_in,
                       float* __restrict__ o,          // [B,T,H*DV]
                       float* __restrict__ state_out) {
    const int bh = blockIdx.x;
    const int b = bh / H_, h = bh % H_;
    const int vc = threadIdx.x;                  // 0..191
    const int wv = threadIdx.x >> 5;

    __shared__ float stage[2][2 * CH * DK];      // [buf][ k rows | q rows ]
    __shared__ float sgst[2][2 * CH];            // [buf][ exp(g) | beta ]

    float S[DK];
    const float* S0 = state_in + (size_t)b * CACHE_ + (size_t)h * DK * DV;
#pragma unroll
    for (int kk = 0; kk < DK; kk++) S[kk] = S0[(size_t)kk * DV + vc];
    const float scale = rsqrtf((float)DK);

    const size_t qkRow = (size_t)H_ * DK;        // row stride in q/k
    const float* kBase = k + (size_t)b * T_ * qkRow + (size_t)h * DK;
    const float* qBase = q + (size_t)b * T_ * qkRow + (size_t)h * DK;

#if HAVE_TDM
    if (wv == 0) {
        tdm_load_2d(kBase, lds_off_of(&stage[0][0]),          DK, CH, (long)qkRow);
        tdm_load_2d(qBase, lds_off_of(&stage[0][CH * DK]),    DK, CH, (long)qkRow);
    }
#endif

    int c = 0;
    for (int t0 = 0; t0 < T_; t0 += CH, c ^= 1) {
        // per-chunk scalars (double-buffered)
        if (vc >= 96 && vc < 96 + CH)
            sgst[c][vc - 96] = __expf(g[((size_t)b * T_ + t0 + (vc - 96)) * H_ + h]);
        if (vc >= 104 && vc < 104 + CH)
            sgst[c][CH + (vc - 104)] = beta[((size_t)b * T_ + t0 + (vc - 104)) * H_ + h];
#if HAVE_TDM
        if (wv == 0) __builtin_amdgcn_s_wait_tensorcnt(0);
#else
        for (int i = vc; i < 2 * CH * DK; i += 192) {
            int half = i / (CH * DK);                     // 0 = k, 1 = q
            int s = (i % (CH * DK)) / DK, kk = i % DK;
            const float* src = half ? qBase : kBase;
            stage[c][i] = src[(size_t)(t0 + s) * qkRow + kk];
        }
#endif
        __syncthreads();
#if HAVE_TDM
        if (wv == 0 && t0 + CH < T_) {
            tdm_load_2d(kBase + (size_t)(t0 + CH) * qkRow, lds_off_of(&stage[c ^ 1][0]),       DK, CH, (long)qkRow);
            tdm_load_2d(qBase + (size_t)(t0 + CH) * qkRow, lds_off_of(&stage[c ^ 1][CH * DK]), DK, CH, (long)qkRow);
        }
#endif
        for (int s = 0; s < CH; s++) {
            const float decay = sgst[c][s];
            const float bta   = sgst[c][CH + s];
            const float* kt = &stage[c][s * DK];
            const float* qt = &stage[c][CH * DK + s * DK];
            const float vt = v[((size_t)b * T_ + t0 + s) * (H_ * DV) + (size_t)h * DV + vc];
            float v_old = 0.f;
#pragma unroll
            for (int kk = 0; kk < DK; kk++) { S[kk] *= decay; v_old = fmaf(kt[kk], S[kk], v_old); }
            const float delta = bta * (vt - v_old);
            float oacc = 0.f;
#pragma unroll
            for (int kk = 0; kk < DK; kk++) { S[kk] = fmaf(kt[kk], delta, S[kk]); oacc = fmaf(qt[kk], S[kk], oacc); }
            o[((size_t)b * T_ + t0 + s) * (H_ * DV) + (size_t)h * DV + vc] = oacc * scale;
        }
    }
    float* Sout = state_out + (size_t)b * CACHE_ + (size_t)h * DK * DV;
#pragma unroll
    for (int kk = 0; kk < DK; kk++) Sout[(size_t)kk * DV + vc] = S[kk];
}

// ---------------- gated RMSNorm epilogue -> bf16 : one wave per (b,t,h) ----------------
__global__ void out_gate_kernel(const float* __restrict__ o,    // [BT*H, DV]
                                const float* __restrict__ gate, // [BT*H, DV]
                                const float* __restrict__ w,    // [DV]
                                __bf16* __restrict__ ob) {      // [BT, VD] bf16
    int gw = (blockIdx.x * blockDim.x + threadIdx.x) >> 5;
    int lane = threadIdx.x & 31;
    if (gw >= BT * H_) return;
    const float* op = o    + (size_t)gw * DV;
    const float* gp = gate + (size_t)gw * DV;
    __bf16* obp = ob + (size_t)gw * DV;
    float xv[6]; float ss = 0.f;
#pragma unroll
    for (int i = 0; i < 6; i++) { xv[i] = op[lane + 32 * i]; ss += xv[i] * xv[i]; }
    for (int m = 16; m; m >>= 1) ss += __shfl_xor(ss, m, 32);
    float r = rsqrtf(ss / (float)DV + 1e-5f);
#pragma unroll
    for (int i = 0; i < 6; i++) {
        int idx = lane + 32 * i;
        float gg = gp[idx];
        float y = xv[i] * r * w[idx] * (gg / (1.f + __expf(-gg)));
        obp[idx] = (__bf16)y;
    }
}

// ---------------- launch ----------------
extern "C" void kernel_launch(void* const* d_in, const int* in_sizes, int n_in,
                              void* d_out, int out_size, void* d_ws, size_t ws_size,
                              hipStream_t stream) {
    const float* x    = (const float*)d_in[0];
    const float* rnn  = (const float*)d_in[1];   // [1,B,CACHE]
    const float* Wq   = (const float*)d_in[2];
    const float* Wk   = (const float*)d_in[3];
    const float* Wv   = (const float*)d_in[4];
    const float* Wa   = (const float*)d_in[5];
    const float* Wb   = (const float*)d_in[6];
    const float* dtb  = (const float*)d_in[7];
    const float* Alog = (const float*)d_in[8];
    const float* Wg   = (const float*)d_in[9];
    const float* cqw  = (const float*)d_in[10];
    const float* ckw  = (const float*)d_in[11];
    const float* cvw  = (const float*)d_in[12];
    const float* onw  = (const float*)d_in[13];
    const float* Wo   = (const float*)d_in[14];

    float* out       = (float*)d_out;             // [B,T,C]
    float* state_out = out + (size_t)BT * C_;     // [B,CACHE]

    // workspace carve-up
    char* ws = (char*)d_ws;
    size_t off = 0;
    auto carve = [&](size_t bytes) { void* p = ws + off; off = (off + bytes + 255) & ~(size_t)255; return p; };
    __bf16* xb   = (__bf16*)carve((size_t)BT * C_ * 2);
    __bf16* WqT  = (__bf16*)carve((size_t)QKD * C_ * 2);
    __bf16* WkT  = (__bf16*)carve((size_t)QKD * C_ * 2);
    __bf16* WvT  = (__bf16*)carve((size_t)VD * C_ * 2);
    __bf16* WgT  = (__bf16*)carve((size_t)VD * C_ * 2);
    __bf16* WoT  = (__bf16*)carve((size_t)C_ * VD * 2);
    float*  qb   = (float*)carve((size_t)BT * QKD * 4);
    float*  kb   = (float*)carve((size_t)BT * QKD * 4);
    float*  vb   = (float*)carve((size_t)BT * VD * 4);
    float*  gateb= (float*)carve((size_t)BT * VD * 4);
    float*  obuf = (float*)carve((size_t)BT * VD * 4);
    float*  gbuf = (float*)carve((size_t)BT * H_ * 4);
    float*  betab= (float*)carve((size_t)BT * H_ * 4);
    __bf16* obf  = (__bf16*)carve((size_t)BT * VD * 2);
    (void)ws_size; (void)n_in; (void)in_sizes; (void)out_size;

    const int TB = 256;
    // 1) bf16 conversions
    cvt_bf16_kernel<<<(BT * C_ + TB - 1) / TB, TB, 0, stream>>>(x, xb, BT * C_);
    transpose_bf16_kernel<<<(C_ * QKD + TB - 1) / TB, TB, 0, stream>>>(Wq, WqT, C_, QKD);
    transpose_bf16_kernel<<<(C_ * QKD + TB - 1) / TB, TB, 0, stream>>>(Wk, WkT, C_, QKD);
    transpose_bf16_kernel<<<(C_ * VD  + TB - 1) / TB, TB, 0, stream>>>(Wv, WvT, C_, VD);
    transpose_bf16_kernel<<<(C_ * VD  + TB - 1) / TB, TB, 0, stream>>>(Wg, WgT, C_, VD);
    transpose_bf16_kernel<<<(VD * C_  + TB - 1) / TB, TB, 0, stream>>>(Wo, WoT, VD, C_);

    // 2) projection GEMMs (WMMA bf16, 32x64 per wave)
    auto gemm = [&](const __bf16* A, const __bf16* Bt, float* Cm, int M, int N, int Kd) {
        long waves = (long)(M >> 5) * (N >> 6);
        int blocks = (int)((waves + 7) / 8);
        wmma_gemm_bf16_kernel<<<blocks, TB, 0, stream>>>(A, Bt, Cm, M, N, Kd);
    };
    gemm(xb, WqT, qb,    BT, QKD, C_);
    gemm(xb, WkT, kb,    BT, QKD, C_);
    gemm(xb, WvT, vb,    BT, VD,  C_);
    gemm(xb, WgT, gateb, BT, VD,  C_);

    // 3) beta / g heads
    ab_head_kernel<<<(BT * H_ * 32 + TB - 1) / TB, TB, 0, stream>>>(x, Wa, Wb, dtb, Alog, gbuf, betab);

    // 4) causal conv + SiLU (in place) + new conv caches
    conv_silu_kernel<<<(B_ * QKD + TB - 1) / TB, TB, 0, stream>>>(qb, cqw, rnn, state_out, QKD, OFF_CQ);
    conv_silu_kernel<<<(B_ * QKD + TB - 1) / TB, TB, 0, stream>>>(kb, ckw, rnn, state_out, QKD, OFF_CK);
    conv_silu_kernel<<<(B_ * VD  + TB - 1) / TB, TB, 0, stream>>>(vb, cvw, rnn, state_out, VD,  OFF_CV);

    // 5) L2 norm q, k
    l2norm96_kernel<<<(BT * H_ * 32 + TB - 1) / TB, TB, 0, stream>>>(qb);
    l2norm96_kernel<<<(BT * H_ * 32 + TB - 1) / TB, TB, 0, stream>>>(kb);

    // 6) recurrence (state in VGPRs, TDM-staged k/q chunks)
    recurrence_kernel<<<B_ * H_, DV, 0, stream>>>(qb, kb, vb, gbuf, betab, rnn, obuf, state_out);

    // 7) gated RMSNorm -> bf16
    out_gate_kernel<<<(BT * H_ * 32 + TB - 1) / TB, TB, 0, stream>>>(obuf, gateb, onw, obf);

    // 8) output projection (WMMA bf16) -> d_out
    gemm(obf, WoT, out, BT, C_, VD);
}
